// PlaneNet_85358180041621
// MI455X (gfx1250) — matrix operand order repaired and checked by reading, not compiled, not run on hardware
//
#include <hip/hip_runtime.h>

// ---------------- problem constants ----------------
#define PP   3
#define NN   50000
#define CC   5
#define FF   68        // features per class
#define EE   150000    // edges per plane
#define EFD  64        // edge hidden
#define NFD  64        // node hidden
#define K1   136       // 2*FF
#define K1P  160       // padded to 5 chunks of 32
#define KC1  5
#define KC2  2         // 64 / 32
#define NT   4         // 64 / 16

typedef __attribute__((ext_vector_type(16))) _Float16 v16h;
typedef __attribute__((ext_vector_type(8)))  _Float16 v8h;
typedef __attribute__((ext_vector_type(8)))  float    v8f;

// ---------------- helpers ----------------
static __device__ inline v8f wmma_f16(v16h a, v16h b, v8f c) {
  return __builtin_amdgcn_wmma_f32_16x16x32_f16(
      /*neg_a=*/false, a, /*neg_b=*/false, b,
      /*c_mod=*/(short)0, c, /*reuse_a=*/false, /*reuse_b=*/false);
}

static __device__ inline float fast_tanh(float x) {
#if __has_builtin(__builtin_amdgcn_tanhf)
  return __builtin_amdgcn_tanhf(x);
#else
  float r;
  asm("v_tanh_f32 %0, %1" : "=v"(r) : "v"(x));
  return r;
#endif
}

// Build a 16-half A fragment from two contiguous 16B runs (K .. K+7, K+16 .. K+23)
static __device__ inline v16h load_a_frag(const _Float16* row) {
  v8h lo = *reinterpret_cast<const v8h*>(row);
  v8h hi = *reinterpret_cast<const v8h*>(row + 16);
  return __builtin_shufflevector(lo, hi, 0,1,2,3,4,5,6,7,8,9,10,11,12,13,14,15);
}

// ---------------- workspace layout (bytes) ----------------
#define AGGR_OFF   0                              // P*N*C*F floats = 204,000,000 B
#define EW1P_OFF   204000000                      // 3*5*5*4*512 halves = 307,200 B
#define NW1P_OFF   204307200                      // 307,200 B
#define NW2P_OFF   204614400                      // 3*5*2*4*512 halves = 122,880 B
#define WS_NEEDED  204737280

// ---------------- kernel: zero aggr ----------------
__global__ void zero_aggr_kernel(float4* __restrict__ p, int n4) {
  int i = blockIdx.x * blockDim.x + threadIdx.x;
  if (i < n4) p[i] = make_float4(0.f, 0.f, 0.f, 0.f);
}

// ---------------- kernel: pack weights to f16 B-fragment layout ----------------
// layout: [(p,c,kc,nt) block of 512 halves][lane*16 + j]
// element (lane,j): K = kc*32 + (lane>=16 ? 8 : 0) + (j&7) + (j>=8 ? 16 : 0) ; n = nt*16 + lane%16
static __device__ inline void pack_one(const float* __restrict__ src,
                                       _Float16* __restrict__ dst,
                                       int idx, int KC, int Kdim) {
  int j    = idx & 15;
  int lane = (idx >> 4) & 31;
  int rest = idx >> 9;
  int nt = rest & 3; rest >>= 2;
  int kc = rest % KC; rest /= KC;
  int c  = rest % CC;
  int p  = rest / CC;
  int K  = kc * 32 + ((lane >= 16) ? 8 : 0) + (j & 7) + ((j >> 3) << 4);
  int n  = nt * 16 + (lane & 15);
  float v = (K < Kdim) ? src[(((size_t)p * CC + c) * Kdim + K) * 64 + n] : 0.f;
  dst[idx] = (_Float16)v;
}

__global__ void pack_weights_kernel(const float* __restrict__ ew1,
                                    const float* __restrict__ nw1,
                                    const float* __restrict__ nw2,
                                    _Float16* __restrict__ ew1p,
                                    _Float16* __restrict__ nw1p,
                                    _Float16* __restrict__ nw2p) {
  const int T1 = PP * CC * KC1 * NT * 512;   // 153600
  const int T2 = 2 * T1;                     // 307200
  const int T3 = T2 + PP * CC * KC2 * NT * 512; // 368640
  int tid = blockIdx.x * blockDim.x + threadIdx.x;
  if (tid < T1)       pack_one(ew1, ew1p, tid,        KC1, K1);
  else if (tid < T2)  pack_one(nw1, nw1p, tid - T1,   KC1, K1);
  else if (tid < T3)  pack_one(nw2, nw2p, tid - T2,   KC2, 64);
}

// ---------------- kernel: edge MLP + softmax attention + scatter-add ----------------
__global__ __launch_bounds__(64) void edge_kernel(
    const float* __restrict__ x,
    const long long* __restrict__ ei,
    const _Float16* __restrict__ ew1p,
    const float* __restrict__ eb1,
    const float* __restrict__ ew2,
    const float* __restrict__ eb2,
    float* __restrict__ aggr) {
  __shared__ __align__(16) _Float16 s_ein[2][16][CC][K1P];
  __shared__ __align__(16) float    s_h[2][16][68];       // padded row (bank spread)
  __shared__ float s_w[2][16][CC];
  __shared__ int   s_dst[2][16];
  __shared__ int   s_src[2][16];

  const int lane = threadIdx.x & 31;
  const int w    = threadIdx.x >> 5;
  const int gw   = blockIdx.x * 2 + w;
  const int tilesPerPlane = EE / 16;                       // 9375
  if (gw >= PP * tilesPerPlane) return;
  const int p    = gw / tilesPerPlane;
  const int tile = gw % tilesPerPlane;
  const int e0   = tile * 16;

  // edge indices (src = ei[p][0][:], dst = ei[p][1][:])
  if (lane < 16) {
    s_src[w][lane] = (int)ei[(size_t)p * 2 * EE + (e0 + lane)];
  } else {
    s_dst[w][lane - 16] = (int)ei[(size_t)p * 2 * EE + EE + (e0 + lane - 16)];
  }

  // stage e_in = [x_i(dst) | x_j(src)] as f16, K padded to 160
  for (int idx = lane; idx < 16 * CC * FF; idx += 32) {
    int m = idx / (CC * FF);
    int r = idx - m * (CC * FF);
    int c = r / FF;
    int f = r - c * FF;
    int nd = s_dst[w][m], ns = s_src[w][m];
    float xi = x[(((size_t)p * NN + nd) * CC + c) * FF + f];
    float xj = x[(((size_t)p * NN + ns) * CC + c) * FF + f];
    s_ein[w][m][c][f]      = (_Float16)xi;
    s_ein[w][m][c][FF + f] = (_Float16)xj;
  }
  for (int idx = lane; idx < 16 * CC * (K1P - K1); idx += 32) {
    int m = idx / (CC * (K1P - K1));
    int r = idx - m * (CC * (K1P - K1));
    int c = r / (K1P - K1);
    int q = r - c * (K1P - K1);
    s_ein[w][m][c][K1 + q] = (_Float16)0.f;
  }

  const int ml   = lane & 15;
  const int off8 = (lane >> 4) << 3;  // 0 for lanes 0-15, 8 for lanes 16-31
  const int hi8  = off8;              // C/D row offset

  for (int c = 0; c < CC; ++c) {
    v16h a[KC1];
#pragma unroll
    for (int kc = 0; kc < KC1; ++kc)
      a[kc] = load_a_frag(&s_ein[w][ml][c][kc * 32 + off8]);

#pragma unroll
    for (int nt = 0; nt < NT; ++nt) {
      v8f acc = {};
#pragma unroll
      for (int kc = 0; kc < KC1; ++kc) {
        const v16h b = *reinterpret_cast<const v16h*>(
            ew1p + ((((size_t)p * CC + c) * KC1 + kc) * NT + nt) * 512 + (size_t)lane * 16);
        acc = wmma_f16(a[kc], b, acc);
      }
      float bias = eb1[((size_t)p * CC + c) * EFD + nt * 16 + ml];
#pragma unroll
      for (int i = 0; i < 8; ++i)
        s_h[w][i + hi8][nt * 16 + ml] = fast_tanh(acc[i] + bias);
    }

    // logit[m] = h . ew2 + eb2 ; lane pairs split K
    float sum = 0.f;
    const int kh = (lane >> 4) * 32;
#pragma unroll 8
    for (int k = 0; k < 32; ++k)
      sum += s_h[w][ml][kh + k] * ew2[((size_t)p * CC + c) * EFD + kh + k];
    sum += __shfl_xor(sum, 16, 32);
    if (lane < 16) s_w[w][ml][c] = sum + eb2[p * CC + c];
  }

  // softmax over classes (lanes 0-15, one edge each)
  if (lane < 16) {
    float lg[CC], mx = -1e30f;
#pragma unroll
    for (int c = 0; c < CC; ++c) { lg[c] = s_w[w][ml][c]; mx = fmaxf(mx, lg[c]); }
    float s = 0.f;
#pragma unroll
    for (int c = 0; c < CC; ++c) { lg[c] = __expf(lg[c] - mx); s += lg[c]; }
    float inv = 1.f / s;
#pragma unroll
    for (int c = 0; c < CC; ++c) s_w[w][ml][c] = lg[c] * inv;
  }

  // msg = w * x_j, scatter-add by dst (coalesced over f)
  for (int idx = lane; idx < 16 * CC * FF; idx += 32) {
    int m = idx / (CC * FF);
    int r = idx - m * (CC * FF);
    int c = r / FF;
    int f = r - c * FF;
    if (e0 + m < EE) {
      float v = s_w[w][m][c] * (float)s_ein[w][m][c][FF + f];
      atomicAdd(&aggr[(((size_t)p * NN + s_dst[w][m]) * CC + c) * FF + f], v);
    }
  }
}

// ---------------- kernel: node update MLP (two GEMMs) ----------------
__global__ __launch_bounds__(64) void node_kernel(
    const float* __restrict__ x,
    const float* __restrict__ aggr,
    const _Float16* __restrict__ nw1p,
    const float* __restrict__ nb1,
    const _Float16* __restrict__ nw2p,
    const float* __restrict__ nb2,
    float* __restrict__ out) {
  __shared__ __align__(16) _Float16 s_uin[2][16][CC][K1P];
  __shared__ __align__(16) _Float16 s_h2[2][16][64];

  const int lane = threadIdx.x & 31;
  const int w    = threadIdx.x >> 5;
  const int gw   = blockIdx.x * 2 + w;
  const int tilesPerPlane = NN / 16;                       // 3125
  if (gw >= PP * tilesPerPlane) return;
  const int p    = gw / tilesPerPlane;
  const int tile = gw % tilesPerPlane;
  const int n0   = tile * 16;

  // stage u_in = [x | aggr], fully coalesced (consecutive nodes)
  for (int idx = lane; idx < 16 * CC * FF; idx += 32) {
    int m = idx / (CC * FF);
    int r = idx - m * (CC * FF);
    int c = r / FF;
    int f = r - c * FF;
    size_t base = (((size_t)p * NN + n0 + m) * CC + c) * FF + f;
    s_uin[w][m][c][f]      = (_Float16)x[base];
    s_uin[w][m][c][FF + f] = (_Float16)aggr[base];
  }
  for (int idx = lane; idx < 16 * CC * (K1P - K1); idx += 32) {
    int m = idx / (CC * (K1P - K1));
    int r = idx - m * (CC * (K1P - K1));
    int c = r / (K1P - K1);
    int q = r - c * (K1P - K1);
    s_uin[w][m][c][K1 + q] = (_Float16)0.f;
  }

  const int ml   = lane & 15;
  const int off8 = (lane >> 4) << 3;
  const int hi8  = off8;

  for (int c = 0; c < CC; ++c) {
    v16h a[KC1];
#pragma unroll
    for (int kc = 0; kc < KC1; ++kc)
      a[kc] = load_a_frag(&s_uin[w][ml][c][kc * 32 + off8]);

    // GEMM1: h2 = tanh(u_in @ nw1 + nb1) -> LDS (f16, A-friendly row-major)
#pragma unroll
    for (int nt = 0; nt < NT; ++nt) {
      v8f acc = {};
#pragma unroll
      for (int kc = 0; kc < KC1; ++kc) {
        const v16h b = *reinterpret_cast<const v16h*>(
            nw1p + ((((size_t)p * CC + c) * KC1 + kc) * NT + nt) * 512 + (size_t)lane * 16);
        acc = wmma_f16(a[kc], b, acc);
      }
      float bias = nb1[((size_t)p * CC + c) * NFD + nt * 16 + ml];
#pragma unroll
      for (int i = 0; i < 8; ++i)
        s_h2[w][i + hi8][nt * 16 + ml] = (_Float16)fast_tanh(acc[i] + bias);
    }

    // GEMM2: out = tanh(h2 @ nw2 + nb2)
    v16h a2[KC2];
#pragma unroll
    for (int kc = 0; kc < KC2; ++kc)
      a2[kc] = load_a_frag(&s_h2[w][ml][kc * 32 + off8]);

#pragma unroll
    for (int nt = 0; nt < NT; ++nt) {
      v8f acc = {};
#pragma unroll
      for (int kc = 0; kc < KC2; ++kc) {
        const v16h b = *reinterpret_cast<const v16h*>(
            nw2p + ((((size_t)p * CC + c) * KC2 + kc) * NT + nt) * 512 + (size_t)lane * 16);
        acc = wmma_f16(a2[kc], b, acc);
      }
      float bias = nb2[((size_t)p * CC + c) * NFD + nt * 16 + ml];
#pragma unroll
      for (int i = 0; i < 8; ++i)
        out[(((size_t)p * NN + n0 + i + hi8) * CC + c) * NFD + nt * 16 + ml] =
            fast_tanh(acc[i] + bias);
    }
  }
}

// ---------------- host launcher ----------------
extern "C" void kernel_launch(void* const* d_in, const int* in_sizes, int n_in,
                              void* d_out, int out_size, void* d_ws, size_t ws_size,
                              hipStream_t stream) {
  (void)in_sizes; (void)n_in; (void)out_size;
  if (ws_size < (size_t)WS_NEEDED) return;

  const float*     x    = (const float*)d_in[0];
  const long long* ei   = (const long long*)d_in[1];
  const float*     ew1  = (const float*)d_in[2];
  const float*     eb1  = (const float*)d_in[3];
  const float*     ew2  = (const float*)d_in[4];
  const float*     eb2  = (const float*)d_in[5];
  const float*     nw1  = (const float*)d_in[6];
  const float*     nb1  = (const float*)d_in[7];
  const float*     nw2  = (const float*)d_in[8];
  const float*     nb2  = (const float*)d_in[9];
  float*           out  = (float*)d_out;

  char* wsb = (char*)d_ws;
  float*     aggr = (float*)(wsb + AGGR_OFF);
  _Float16*  ew1p = (_Float16*)(wsb + EW1P_OFF);
  _Float16*  nw1p = (_Float16*)(wsb + NW1P_OFF);
  _Float16*  nw2p = (_Float16*)(wsb + NW2P_OFF);

  // 1) zero the scatter-add accumulator (P*N*C*F floats, as float4)
  {
    int n4 = (PP * NN * CC * FF) / 4;  // 12,750,000
    int blk = 256;
    int grid = (n4 + blk - 1) / blk;
    zero_aggr_kernel<<<grid, blk, 0, stream>>>((float4*)aggr, n4);
  }
  // 2) repack weights into f16 WMMA B fragments
  {
    int total = PP * CC * NT * 512 * (KC1 + KC1 + KC2);  // 368,640
    int blk = 256;
    int grid = (total + blk - 1) / blk;
    pack_weights_kernel<<<grid, blk, 0, stream>>>(ew1, nw1, nw2, ew1p, nw1p, nw2p);
  }
  // 3) edge MLP + attention + scatter
  {
    int waves = PP * (EE / 16);                // 28,125
    int grid = (waves + 1) / 2;                // 2 waves / block
    edge_kernel<<<grid, 64, 0, stream>>>(x, ei, ew1p, eb1, ew2, eb2, aggr);
  }
  // 4) node MLP
  {
    int waves = PP * (NN / 16);                // 9,375
    int grid = (waves + 1) / 2;
    node_kernel<<<grid, 64, 0, stream>>>(x, aggr, nw1p, nb1, nw2p, nb2, out);
  }
}
